// VectorQuantize_34222299415125
// MI455X (gfx1250) — matrix operand order repaired and checked by reading, not compiled, not run on hardware
//
#include <hip/hip_runtime.h>

// ---------------------------------------------------------------------------
// VectorQuantize for MI455X (gfx1250): bf16 WMMA pipeline, async-LDS staging
//   B=8 N=2048 DIM=1024 HEADS=8 CB_DIM=64 CB_SIZE=1024 CB_IN=512
// ---------------------------------------------------------------------------

typedef __attribute__((ext_vector_type(16))) __bf16 v16bf;
typedef __attribute__((ext_vector_type(8)))  float  v8f;

union FragU { v16bf v; unsigned int u[8]; };
union AccU  { v8f   v; float        f[8]; };

__device__ __forceinline__ unsigned short f2bf(float f) {
    union { float f; unsigned int u; } x; x.f = f;
    unsigned int r = x.u + 0x7FFFu + ((x.u >> 16) & 1u);   // round-to-nearest-even
    return (unsigned short)(r >> 16);
}

// 16-bit A-matrix 16x32 K-pair base for VGPR i (0..7), hi = lane>=16
__device__ __forceinline__ int kmapA(int i, int hi) {
    return (i < 4) ? (2 * i + 8 * hi) : (16 + 2 * (i - 4) + 8 * hi);
}
// 16-bit B-matrix 32x16: same K-pair order as A (table order)
__device__ __forceinline__ int kmapB(int i, int hi) {
    return (i < 4) ? (2 * i + 8 * hi) : (16 + 2 * (i - 4) + 8 * hi);
}

// ---- gfx1250 async global->LDS (ASYNCcnt) with safe fallback ---------------
#if defined(__has_builtin)
#  if __has_builtin(__builtin_amdgcn_global_load_async_to_lds_b128)
#    define ATHENA_ASYNC_LDS 1
#  endif
#endif

typedef int v4i_ty __attribute__((vector_size(16)));
typedef v4i_ty __attribute__((address_space(1)))* as1_v4i_ptr;
typedef v4i_ty __attribute__((address_space(3)))* as3_v4i_ptr;

__device__ __forceinline__ void async_copy16(const void* gsrc, void* lds_dst) {
#ifdef ATHENA_ASYNC_LDS
    __builtin_amdgcn_global_load_async_to_lds_b128(
        (as1_v4i_ptr)gsrc, (as3_v4i_ptr)lds_dst, 0, 0);
#else
    *(uint4*)lds_dst = *(const uint4*)gsrc;
#endif
}
__device__ __forceinline__ void async_wait_all() {
#ifdef ATHENA_ASYNC_LDS
    asm volatile("s_wait_asynccnt 0x0" ::: "memory");
#endif
}

// ---------------------------------------------------------------------------
// f32 -> bf16 convert
// ---------------------------------------------------------------------------
__global__ __launch_bounds__(256) void cvt_bf16_kernel(
    const float* __restrict__ src, unsigned short* __restrict__ dst, int n) {
    int i = blockIdx.x * 256 + threadIdx.x;
    if (i < n) dst[i] = f2bf(src[i]);
}

// ||e_c||^2 for each code
__global__ __launch_bounds__(256) void esq_kernel(
    const float* __restrict__ embed, float* __restrict__ esq) {
    int c = blockIdx.x * 256 + threadIdx.x;
    if (c < 1024) {
        float s = 0.f;
        #pragma unroll 8
        for (int k = 0; k < 64; ++k) { float e = embed[c * 64 + k]; s += e * e; }
        esq[c] = s;
    }
}

// ---------------------------------------------------------------------------
// Tiled bf16 GEMM: C[M,N] = A[M,K] * B[K,N] + bias
//   block = 256 threads (8 wave32), tile = 128(M) x 64(N), K step 32
//   4 x v_wmma_f32_16x16x32_bf16 per wave per K-step, double-buffered LDS,
//   A staged via GLOBAL_LOAD_ASYNC_TO_LDS_B128 when available.
//   GATHER_A: A rows are embed[ gidx[row*8 + k/64] ][k%64]  (fused VQ gather)
// ---------------------------------------------------------------------------
template <bool OUT_BF16, bool GATHER_A>
__global__ __launch_bounds__(256) void gemm_bf16_kernel(
    const unsigned short* __restrict__ A,   // or embed table [1024,64] if GATHER_A
    const int* __restrict__ gidx,           // [M*8] codes if GATHER_A, else unused
    const unsigned short* __restrict__ B,
    const float* __restrict__ bias, void* __restrict__ C,
    int M, int N, int K) {
    __shared__ unsigned short As[2][128 * 40];  // 128 rows x 32 k, stride 40 (80B)
    __shared__ unsigned short Bs[2][64 * 40];   // transposed: 64 cols x 32 k

    const int tid  = threadIdx.x;
    const int lane = tid & 31;
    const int wave = tid >> 5;
    const int hi   = lane >> 4;
    const int ln   = lane & 15;
    const int bm   = blockIdx.y * 128;
    const int bn   = blockIdx.x * 64;
    (void)M;

    AccU acc[4];
    #pragma unroll
    for (int s = 0; s < 4; ++s)
        #pragma unroll
        for (int i = 0; i < 8; ++i) acc[s].f[i] = 0.f;

    // --- staging helpers -----------------------------------------------
    auto stageA = [&](int buf, int k0) {
        #pragma unroll
        for (int rep = 0; rep < 2; ++rep) {
            int idx = tid + rep * 256;          // 0..511
            int r   = idx >> 2;                 // row 0..127
            int c4  = idx & 3;                  // 8-elem chunk
            const unsigned short* src;
            if (GATHER_A) {
                int kcol = k0 + c4 * 8;
                int head = kcol >> 6;           // CB_DIM = 64
                int code = gidx[(size_t)(bm + r) * 8 + head];
                src = A + (size_t)code * 64 + (kcol & 63);
            } else {
                src = A + (size_t)(bm + r) * K + k0 + c4 * 8;
            }
            async_copy16(src, &As[buf][r * 40 + c4 * 8]);
        }
    };
    auto stageB = [&](int buf, int k0) {
        #pragma unroll
        for (int rep = 0; rep < 8; ++rep) {
            int idx = tid + rep * 256;          // 0..2047
            int kk  = idx >> 6;                 // 0..31
            int nn  = idx & 63;
            Bs[buf][nn * 40 + kk] = B[(size_t)(k0 + kk) * N + bn + nn];
        }
    };

    stageA(0, 0);
    stageB(0, 0);

    const int nk = K >> 5;
    for (int kt = 0; kt < nk; ++kt) {
        const int p = kt & 1;
        async_wait_all();          // this wave's async loads into buf p complete
        __syncthreads();           // everyone's loads + ds stores visible

        if (kt + 1 < nk) {         // prefetch next K-slab into the other buffer
            stageA(p ^ 1, (kt + 1) * 32);
            stageB(p ^ 1, (kt + 1) * 32);
        }

        FragU fa;
        #pragma unroll
        for (int i = 0; i < 8; ++i)
            fa.u[i] = *(const unsigned int*)(&As[p][(wave * 16 + ln) * 40 + kmapA(i, hi)]);

        #pragma unroll
        for (int s = 0; s < 4; ++s) {
            FragU fb;
            #pragma unroll
            for (int i = 0; i < 8; ++i)
                fb.u[i] = *(const unsigned int*)(&Bs[p][(s * 16 + ln) * 40 + kmapB(i, hi)]);
            acc[s].v = __builtin_amdgcn_wmma_f32_16x16x32_bf16(
                false, fa.v, false, fb.v, (short)0, acc[s].v, false, false);
        }
    }

    // Epilogue: D layout — vgpr v: lanes0-15 -> M=v, lanes16-31 -> M=v+8; N = lane&15
    #pragma unroll
    for (int s = 0; s < 4; ++s) {
        const int col = bn + s * 16 + ln;
        const float bv = bias ? bias[col] : 0.f;
        #pragma unroll
        for (int v = 0; v < 8; ++v) {
            int row = bm + wave * 16 + v + 8 * hi;
            float val = acc[s].f[v] + bv;
            if (OUT_BF16) ((unsigned short*)C)[(size_t)row * N + col] = f2bf(val);
            else          ((float*)C)[(size_t)row * N + col] = val;
        }
    }
}

// ---------------------------------------------------------------------------
// Score + argmax: rows = (token,head) pairs (131072), K=64, codes=1024
//   score = 2*(h . e_c) - ||e_c||^2 ; running argmax fused into WMMA loop
// ---------------------------------------------------------------------------
__global__ __launch_bounds__(256) void score_argmax_kernel(
    const unsigned short* __restrict__ H,   // [Mrows,64] bf16
    const unsigned short* __restrict__ E,   // [1024,64] bf16 (code-major)
    const float* __restrict__ esq,          // [1024]
    int* __restrict__ ind, int Mrows) {
    const int tid  = threadIdx.x;
    const int lane = tid & 31;
    const int wave = tid >> 5;
    const int hi   = lane >> 4;
    const int ln   = lane & 15;
    const int rowbase = blockIdx.x * 128 + wave * 16;
    (void)Mrows;

    // A fragments for both K-steps, kept in registers for the whole code loop
    FragU fa0, fa1;
    const unsigned short* arow = H + (size_t)(rowbase + ln) * 64;
    #pragma unroll
    for (int i = 0; i < 8; ++i) {
        fa0.u[i] = *(const unsigned int*)(arow + kmapA(i, hi));
        fa1.u[i] = *(const unsigned int*)(arow + 32 + kmapA(i, hi));
    }

    float best[8]; int bidx[8];
    #pragma unroll
    for (int v = 0; v < 8; ++v) { best[v] = -3.4e38f; bidx[v] = 0; }

    for (int t = 0; t < 1024; t += 16) {
        const int code = t + ln;                       // this lane's column
        const unsigned short* eb = E + (size_t)code * 64;
        FragU fb0, fb1;
        #pragma unroll
        for (int i = 0; i < 8; ++i) {
            int kb = kmapB(i, hi);
            fb0.u[i] = *(const unsigned int*)(eb + kb);
            fb1.u[i] = *(const unsigned int*)(eb + 32 + kb);
        }
        AccU c;
        #pragma unroll
        for (int i = 0; i < 8; ++i) c.f[i] = 0.f;
        c.v = __builtin_amdgcn_wmma_f32_16x16x32_bf16(false, fa0.v, false, fb0.v, (short)0, c.v, false, false);
        c.v = __builtin_amdgcn_wmma_f32_16x16x32_bf16(false, fa1.v, false, fb1.v, (short)0, c.v, false, false);

        const float eq = esq[code];
        #pragma unroll
        for (int v = 0; v < 8; ++v) {
            float s = 2.f * c.f[v] - eq;
            if (s > best[v]) { best[v] = s; bidx[v] = code; }   // strict > keeps lowest idx
        }
    }

    // reduce over the 16 lanes holding one row (xor stays inside each half)
    #pragma unroll
    for (int off = 8; off >= 1; off >>= 1) {
        #pragma unroll
        for (int v = 0; v < 8; ++v) {
            float ob = __shfl_xor(best[v], off, 32);
            int   oi = __shfl_xor(bidx[v], off, 32);
            if (ob > best[v] || (ob == best[v] && oi < bidx[v])) { best[v] = ob; bidx[v] = oi; }
        }
    }
    if (ln == 0) {
        #pragma unroll
        for (int v = 0; v < 8; ++v) ind[rowbase + v + 8 * hi] = bidx[v];
    }
}

// ---------------------------------------------------------------------------
extern "C" void kernel_launch(void* const* d_in, const int* in_sizes, int n_in,
                              void* d_out, int out_size, void* d_ws, size_t ws_size,
                              hipStream_t stream) {
    (void)in_sizes; (void)n_in; (void)out_size; (void)ws_size;
    const int Bv = 8, Nv = 2048, DIM = 1024, HEADS = 8, CBD = 64, CBS = 1024, CBIN = 512;
    const int M  = Bv * Nv;              // 16384 tokens
    const int MH = M * HEADS;            // 131072 (token,head) rows

    const float* x     = (const float*)d_in[0];
    const float* W_in  = (const float*)d_in[1];
    const float* b_in  = (const float*)d_in[2];
    const float* W_out = (const float*)d_in[3];
    const float* b_out = (const float*)d_in[4];
    const float* embed = (const float*)d_in[5];

    float* out = (float*)d_out;                       // [16384,1024] f32
    int*   ind = (int*)d_out + (size_t)M * DIM;       // [131072] int32

    char* ws = (char*)d_ws;
    unsigned short* xb  = (unsigned short*)(ws);                     // 33554432 B
    unsigned short* hb  = (unsigned short*)(ws + 33554432);          // 16777216 B
    unsigned short* wib = (unsigned short*)(ws + 50331648);          //  1048576 B
    unsigned short* wob = (unsigned short*)(ws + 51380224);          //  1048576 B
    unsigned short* eb  = (unsigned short*)(ws + 52428800);          //   131072 B
    float*          esq = (float*)(ws + 52559872);                   //     4096 B

    // 1) convert inputs to bf16 (x stays resident in L2 for the GEMM)
    cvt_bf16_kernel<<<(M * DIM + 255) / 256, 256, 0, stream>>>(x, xb, M * DIM);
    cvt_bf16_kernel<<<(DIM * CBIN + 255) / 256, 256, 0, stream>>>(W_in, wib, DIM * CBIN);
    cvt_bf16_kernel<<<(CBIN * DIM + 255) / 256, 256, 0, stream>>>(W_out, wob, CBIN * DIM);
    cvt_bf16_kernel<<<(CBS * CBD + 255) / 256, 256, 0, stream>>>(embed, eb, CBS * CBD);
    esq_kernel<<<(CBS + 255) / 256, 256, 0, stream>>>(embed, esq);

    // 2) h = x @ W_in + b_in   (bf16 out, f32 accum)
    {
        dim3 grid(CBIN / 64, M / 128);
        gemm_bf16_kernel<true, false><<<grid, 256, 0, stream>>>(
            xb, nullptr, wib, b_in, hb, M, CBIN, DIM);
    }

    // 3) nearest-code argmax per (token, head)
    score_argmax_kernel<<<MH / 128, 256, 0, stream>>>(hb, eb, esq, ind, MH);

    // 4) out = embed[ind] @ W_out + b_out (gather fused into A-staging, f32 out)
    {
        dim3 grid(DIM / 64, M / 128);
        gemm_bf16_kernel<false, true><<<grid, 256, 0, stream>>>(
            eb, ind, wob, b_out, out, M, DIM, CBIN);
    }
}
